// Qwen3VisionAttention_73727408603142
// MI455X (gfx1250) — compile-verified, matrix operand order checked by reading, not run on hardware
//
#include <hip/hip_runtime.h>
#include <hip/hip_bf16.h>

// ---------------------------------------------------------------------------
// Qwen3 vision attention block for gfx1250 (MI455X), bf16x3 WMMA everywhere.
// ---------------------------------------------------------------------------

typedef __bf16 bf16_t;
typedef __attribute__((ext_vector_type(16))) __bf16 v16bf;
typedef __attribute__((ext_vector_type(8)))  float  v8f;

#define S_LEN 3072
#define E_DIM 1280
#define H_NUM 16
#define D_DIM 80
#define DP    96      // padded head dim: 3 * 32
#define NSEG  9       // cu_seqlens entries

__device__ __forceinline__ v8f wmma_bf(v16bf a, v16bf b, v8f c) {
  return __builtin_amdgcn_wmma_f32_16x16x32_bf16(false, a, false, b, (short)0, c,
                                                 false, false);
}

__device__ __forceinline__ v8f zero8() {
  v8f z;
#pragma unroll
  for (int i = 0; i < 8; ++i) z[i] = 0.0f;
  return z;
}

// Load a 16x32 bf16 fragment (A-style / K-contiguous-operand layout per CDNA5
// ISA 7.12.2): lane L holds row (L&15); K chunks {b..b+7, b+16..b+23} with
// b = (L>=16 ? 8 : 0).  Row-major [rows][ld] source (global or LDS).
// Requires: base 16B-aligned, ld % 8 == 0, kbase % 8 == 0  -> two b128 loads.
union FragCast { v16bf f; uint4 q[2]; };

__device__ __forceinline__ v16bf load_frag(const bf16_t* __restrict__ p,
                                           int row_base, int ld, int kbase) {
  const int lane = threadIdx.x & 31;
  const bf16_t* p0 = p + (size_t)(row_base + (lane & 15)) * ld + kbase +
                     ((lane >> 4) << 3);
  FragCast u;
  u.q[0] = *(const uint4*)p0;
  u.q[1] = *(const uint4*)(p0 + 16);
  return u.f;
}

// ---------------------------------------------------------------------------
// fp32 -> (bf16 hi, bf16 lo) split
// ---------------------------------------------------------------------------
__global__ __launch_bounds__(256) void split_bf16x2_kernel(
    const float* __restrict__ in, bf16_t* __restrict__ hi,
    bf16_t* __restrict__ lo, int n) {
  int i = blockIdx.x * 256 + threadIdx.x;
  if (i >= n) return;
  float v = in[i];
  bf16_t h = (bf16_t)v;
  hi[i] = h;
  lo[i] = (bf16_t)(v - (float)h);
}

// ---------------------------------------------------------------------------
// C[M,N] = A[M,K] @ B[N,K]^T + bias, bf16x3 emulated-fp32 WMMA GEMM.
// Block: 256 threads = 8 waves, tile 64(M) x 256(N), K-step 32.
// Wave (wm,wn) computes 32x64 via 2x4 16x16 accumulators.
// ---------------------------------------------------------------------------
__global__ __launch_bounds__(256) void gemm_bf16x3_kernel(
    const bf16_t* __restrict__ Ahi, const bf16_t* __restrict__ Alo,
    const bf16_t* __restrict__ Bhi, const bf16_t* __restrict__ Blo,
    const float* __restrict__ bias, float* __restrict__ C,
    int M, int N, int K) {
  __shared__ __align__(16) bf16_t sAhi[64][32];
  __shared__ __align__(16) bf16_t sAlo[64][32];
  __shared__ __align__(16) bf16_t sBhi[256][32];
  __shared__ __align__(16) bf16_t sBlo[256][32];

  const int tid  = threadIdx.x;
  const int lane = tid & 31;
  const int wave = tid >> 5;
  const int wm   = wave >> 2;  // 0..1
  const int wn   = wave & 3;   // 0..3
  const int n0   = blockIdx.x * 256;
  const int m0   = blockIdx.y * 64;

  v8f acc[2][4];
#pragma unroll
  for (int mi = 0; mi < 2; ++mi)
#pragma unroll
    for (int ni = 0; ni < 4; ++ni) acc[mi][ni] = zero8();

  const int ar  = tid >> 2;          // A stage: row per 4 threads
  const int akk = (tid & 3) << 3;    // 8-elem chunk

  for (int k0 = 0; k0 < K; k0 += 32) {
    // Stage A tile 64x32 (hi+lo): 2048 elems, one 16B chunk per thread.
    *(uint4*)&sAhi[ar][akk] =
        *(const uint4*)(Ahi + (size_t)(m0 + ar) * K + k0 + akk);
    *(uint4*)&sAlo[ar][akk] =
        *(const uint4*)(Alo + (size_t)(m0 + ar) * K + k0 + akk);
    // Stage B tile 256x32 (hi+lo): one row per thread.
#pragma unroll
    for (int j = 0; j < 4; ++j) {
      *(uint4*)&sBhi[tid][j * 8] =
          *(const uint4*)(Bhi + (size_t)(n0 + tid) * K + k0 + j * 8);
      *(uint4*)&sBlo[tid][j * 8] =
          *(const uint4*)(Blo + (size_t)(n0 + tid) * K + k0 + j * 8);
    }
    // Prefetch next K-tile into cache (global_prefetch_b8) behind the WMMAs.
    if (k0 + 32 < K) {
      __builtin_prefetch(Ahi + (size_t)(m0 + ar) * K + k0 + 32 + akk, 0, 1);
      __builtin_prefetch(Alo + (size_t)(m0 + ar) * K + k0 + 32 + akk, 0, 1);
      __builtin_prefetch(Bhi + (size_t)(n0 + tid) * K + k0 + 32, 0, 1);
      __builtin_prefetch(Blo + (size_t)(n0 + tid) * K + k0 + 32, 0, 1);
    }
    __syncthreads();

    v16bf ah[2], al[2];
#pragma unroll
    for (int mi = 0; mi < 2; ++mi) {
      ah[mi] = load_frag(&sAhi[0][0], wm * 32 + mi * 16, 32, 0);
      al[mi] = load_frag(&sAlo[0][0], wm * 32 + mi * 16, 32, 0);
    }
#pragma unroll
    for (int ni = 0; ni < 4; ++ni) {
      v16bf bh = load_frag(&sBhi[0][0], wn * 64 + ni * 16, 32, 0);
      v16bf bl = load_frag(&sBlo[0][0], wn * 64 + ni * 16, 32, 0);
#pragma unroll
      for (int mi = 0; mi < 2; ++mi) {
        acc[mi][ni] = wmma_bf(al[mi], bh, acc[mi][ni]);
        acc[mi][ni] = wmma_bf(ah[mi], bl, acc[mi][ni]);
        acc[mi][ni] = wmma_bf(ah[mi], bh, acc[mi][ni]);
      }
    }
    __syncthreads();
  }

  // Epilogue: C layout VGPR j -> row j (lanes 0-15) / j+8 (lanes 16-31).
  const int rsel = (lane >> 4) << 3;
#pragma unroll
  for (int mi = 0; mi < 2; ++mi)
#pragma unroll
    for (int ni = 0; ni < 4; ++ni)
#pragma unroll
      for (int j = 0; j < 8; ++j) {
        int row = m0 + wm * 32 + mi * 16 + j + rsel;
        int col = n0 + wn * 64 + ni * 16 + (lane & 15);
        C[(size_t)row * N + col] = acc[mi][ni][j] + bias[col];
      }
}

// ---------------------------------------------------------------------------
// RoPE + layout transform:
//   qkv_f32[S][3840] -> q,k : [H][S][DP] bf16 hi/lo (rot applied, padded)
//                      v    : [H][DP][S] bf16 hi/lo (transposed, padded)
// ---------------------------------------------------------------------------
__global__ __launch_bounds__(256) void rope_split_kernel(
    const float* __restrict__ qkv, const float* __restrict__ cosb,
    const float* __restrict__ sinb,
    bf16_t* __restrict__ qhi, bf16_t* __restrict__ qlo,
    bf16_t* __restrict__ khi, bf16_t* __restrict__ klo,
    bf16_t* __restrict__ vthi, bf16_t* __restrict__ vtlo) {
  int idx = blockIdx.x * 256 + threadIdx.x;
  if (idx >= S_LEN * H_NUM * DP) return;
  int d = idx % DP;
  int h = (idx / DP) % H_NUM;
  int s = idx / (DP * H_NUM);
  size_t qk_off = (size_t)h * S_LEN * DP + (size_t)s * DP + d;
  size_t vt_off = (size_t)h * DP * S_LEN + (size_t)d * S_LEN + s;
  if (d >= D_DIM) {
    qhi[qk_off] = (bf16_t)0.f; qlo[qk_off] = (bf16_t)0.f;
    khi[qk_off] = (bf16_t)0.f; klo[qk_off] = (bf16_t)0.f;
    vthi[vt_off] = (bf16_t)0.f; vtlo[vt_off] = (bf16_t)0.f;
    return;
  }
  const float* row = qkv + (size_t)s * (3 * E_DIM);
  float qv = row[h * D_DIM + d];
  float kv = row[E_DIM + h * D_DIM + d];
  float vv = row[2 * E_DIM + h * D_DIM + d];
  float c  = cosb[s * D_DIM + d];
  float sn = sinb[s * D_DIM + d];
  float qr = (d < D_DIM / 2) ? -row[h * D_DIM + d + D_DIM / 2]
                             :  row[h * D_DIM + d - D_DIM / 2];
  float kr = (d < D_DIM / 2) ? -row[E_DIM + h * D_DIM + d + D_DIM / 2]
                             :  row[E_DIM + h * D_DIM + d - D_DIM / 2];
  float qo = qv * c + qr * sn;
  float ko = kv * c + kr * sn;
  bf16_t t;
  t = (bf16_t)qo; qhi[qk_off] = t; qlo[qk_off] = (bf16_t)(qo - (float)t);
  t = (bf16_t)ko; khi[qk_off] = t; klo[qk_off] = (bf16_t)(ko - (float)t);
  t = (bf16_t)vv; vthi[vt_off] = t; vtlo[vt_off] = (bf16_t)(vv - (float)t);
}

// ---------------------------------------------------------------------------
// Flash attention: one wave per (head, 32-row q-block). Online softmax on
// WMMA C fragments; P re-fragmented through LDS; O = 2x6 f32 fragments.
// k-loop starts at kstart & ~31 so every fragment load is 16B-aligned; the
// leading out-of-segment columns are killed by the segment-equality mask.
// ---------------------------------------------------------------------------
#define ATT_WAVES 4

__device__ __forceinline__ int seg_of(const int* cu, int pos) {
  int s = 0;
#pragma unroll
  for (int i = 0; i < NSEG; ++i) s += (cu[i] <= pos) ? 1 : 0;
  return s;
}

__global__ __launch_bounds__(32 * ATT_WAVES) void attn_flash_kernel(
    const bf16_t* __restrict__ qhi, const bf16_t* __restrict__ qlo,
    const bf16_t* __restrict__ khi, const bf16_t* __restrict__ klo,
    const bf16_t* __restrict__ vthi, const bf16_t* __restrict__ vtlo,
    const int* __restrict__ cu, float* __restrict__ ctx) {
  __shared__ __align__(16) bf16_t sPhi[ATT_WAVES][32 * 32];
  __shared__ __align__(16) bf16_t sPlo[ATT_WAVES][32 * 32];

  const int wave = threadIdx.x >> 5;
  const int lane = threadIdx.x & 31;
  const int gw   = blockIdx.x * ATT_WAVES + wave;
  const int h    = gw / (S_LEN / 32);
  const int q0   = (gw % (S_LEN / 32)) * 32;
  const int rsel = (lane >> 4) << 3;

  int cuv[NSEG];
#pragma unroll
  for (int i = 0; i < NSEG; ++i) cuv[i] = cu[i];

  const bf16_t* qh = qhi + (size_t)h * S_LEN * DP;
  const bf16_t* ql = qlo + (size_t)h * S_LEN * DP;
  const bf16_t* kh = khi + (size_t)h * S_LEN * DP;
  const bf16_t* kl = klo + (size_t)h * S_LEN * DP;
  const bf16_t* vh = vthi + (size_t)h * DP * S_LEN;
  const bf16_t* vl = vtlo + (size_t)h * DP * S_LEN;

  float mrun[2][8], lrun[2][8];
  int   segq[2][8];
  v8f   O[2][6];
#pragma unroll
  for (int mi = 0; mi < 2; ++mi) {
#pragma unroll
    for (int j = 0; j < 8; ++j) {
      mrun[mi][j] = -3.0e38f;
      lrun[mi][j] = 0.0f;
      segq[mi][j] = seg_of(cuv, q0 + mi * 16 + j + rsel);
    }
#pragma unroll
    for (int td = 0; td < 6; ++td) O[mi][td] = zero8();
  }

  const int sA = seg_of(cuv, q0);
  const int sB = seg_of(cuv, q0 + 31);
  const int kstart = cuv[sA - 1] & ~31;  // align down; mask kills the overlap
  const int kend   = cuv[sB] < S_LEN ? cuv[sB] : S_LEN;
  const float scale = 0.11180339887498949f;  // 1/sqrt(80)

  for (int k0 = kstart; k0 < kend; k0 += 32) {
    // ---- S = Q K^T (bf16x3, K over padded 96 dims) ----
    v8f Sc[2][2];
    Sc[0][0] = zero8(); Sc[0][1] = zero8();
    Sc[1][0] = zero8(); Sc[1][1] = zero8();
#pragma unroll
    for (int c = 0; c < 3; ++c) {
      v16bf bh[2], bl[2];
#pragma unroll
      for (int tn = 0; tn < 2; ++tn) {
        bh[tn] = load_frag(kh, k0 + tn * 16, DP, c * 32);
        bl[tn] = load_frag(kl, k0 + tn * 16, DP, c * 32);
      }
#pragma unroll
      for (int mi = 0; mi < 2; ++mi) {
        v16bf ah = load_frag(qh, q0 + mi * 16, DP, c * 32);
        v16bf al = load_frag(ql, q0 + mi * 16, DP, c * 32);
#pragma unroll
        for (int tn = 0; tn < 2; ++tn) {
          Sc[mi][tn] = wmma_bf(al, bh[tn], Sc[mi][tn]);
          Sc[mi][tn] = wmma_bf(ah, bl[tn], Sc[mi][tn]);
          Sc[mi][tn] = wmma_bf(ah, bh[tn], Sc[mi][tn]);
        }
      }
    }

    // ---- mask + online softmax ----
    int segk[2];
#pragma unroll
    for (int tn = 0; tn < 2; ++tn) {
      int col = k0 + tn * 16 + (lane & 15);
      segk[tn] = (col < kend) ? seg_of(cuv, col) : -1;
    }
#pragma unroll
    for (int mi = 0; mi < 2; ++mi) {
#pragma unroll
      for (int j = 0; j < 8; ++j) {
        float x0 = Sc[mi][0][j] * scale;
        float x1 = Sc[mi][1][j] * scale;
        if (segk[0] != segq[mi][j]) x0 = -3.0e38f;
        if (segk[1] != segq[mi][j]) x1 = -3.0e38f;
        float mx = fmaxf(x0, x1);
#pragma unroll
        for (int off = 1; off < 16; off <<= 1)
          mx = fmaxf(mx, __shfl_xor(mx, off, 32));
        float mnew = fmaxf(mrun[mi][j], mx);
        float alpha = __expf(mrun[mi][j] - mnew);
        float p0 = (x0 <= -1.0e37f) ? 0.0f : __expf(x0 - mnew);
        float p1 = (x1 <= -1.0e37f) ? 0.0f : __expf(x1 - mnew);
        float rs = p0 + p1;
#pragma unroll
        for (int off = 1; off < 16; off <<= 1)
          rs += __shfl_xor(rs, off, 32);
        lrun[mi][j] = lrun[mi][j] * alpha + rs;
        mrun[mi][j] = mnew;
#pragma unroll
        for (int td = 0; td < 6; ++td) O[mi][td][j] *= alpha;
        // stash P (hi/lo split) into LDS for re-fragmentation
        int prow = mi * 16 + j + rsel;
        bf16_t h0 = (bf16_t)p0, h1 = (bf16_t)p1;
        sPhi[wave][prow * 32 + (lane & 15)]      = h0;
        sPhi[wave][prow * 32 + 16 + (lane & 15)] = h1;
        sPlo[wave][prow * 32 + (lane & 15)]      = (bf16_t)(p0 - (float)h0);
        sPlo[wave][prow * 32 + 16 + (lane & 15)] = (bf16_t)(p1 - (float)h1);
      }
    }

    // ---- O += P V  (P from LDS as A-fragments, V^T is K-contiguous) ----
    v16bf pah[2], pal[2];
#pragma unroll
    for (int mi = 0; mi < 2; ++mi) {
      pah[mi] = load_frag(&sPhi[wave][0], mi * 16, 32, 0);
      pal[mi] = load_frag(&sPlo[wave][0], mi * 16, 32, 0);
    }
#pragma unroll
    for (int td = 0; td < 6; ++td) {
      v16bf vbh = load_frag(vh, td * 16, S_LEN, k0);
      v16bf vbl = load_frag(vl, td * 16, S_LEN, k0);
#pragma unroll
      for (int mi = 0; mi < 2; ++mi) {
        O[mi][td] = wmma_bf(pal[mi], vbh, O[mi][td]);
        O[mi][td] = wmma_bf(pah[mi], vbl, O[mi][td]);
        O[mi][td] = wmma_bf(pah[mi], vbh, O[mi][td]);
      }
    }
  }

  // ---- normalize + scatter to ctx[S][E] (drop the d>=80 padding) ----
#pragma unroll
  for (int mi = 0; mi < 2; ++mi)
#pragma unroll
    for (int j = 0; j < 8; ++j) {
      float inv = 1.0f / fmaxf(lrun[mi][j], 1e-30f);
      int row = q0 + mi * 16 + j + rsel;
#pragma unroll
      for (int td = 0; td < 6; ++td) {
        int d = td * 16 + (lane & 15);
        if (d < D_DIM)
          ctx[(size_t)row * E_DIM + h * D_DIM + d] = O[mi][td][j] * inv;
      }
    }
}

// ---------------------------------------------------------------------------
// Host-side orchestration
// ---------------------------------------------------------------------------
extern "C" void kernel_launch(void* const* d_in, const int* in_sizes, int n_in,
                              void* d_out, int out_size, void* d_ws,
                              size_t ws_size, hipStream_t stream) {
  (void)in_sizes; (void)n_in; (void)out_size; (void)ws_size;
  const float* x      = (const float*)d_in[0];
  const int*   cu     = (const int*)  d_in[1];
  const float* cosb   = (const float*)d_in[2];
  const float* sinb   = (const float*)d_in[3];
  const float* qkv_w  = (const float*)d_in[4];
  const float* qkv_b  = (const float*)d_in[5];
  const float* proj_w = (const float*)d_in[6];
  const float* proj_b = (const float*)d_in[7];
  float* out = (float*)d_out;

  char* p = (char*)d_ws;
  auto carve = [&](size_t bytes) -> void* {
    void* r = (void*)p;
    p += ((bytes + 4095) & ~(size_t)4095) + 8192;  // 8KB tail slack per carve
    return r;
  };

  const size_t nX   = (size_t)S_LEN * E_DIM;          // 3.93M
  const size_t nW1  = (size_t)3 * E_DIM * E_DIM;      // 4.92M
  const size_t nW2  = (size_t)E_DIM * E_DIM;          // 1.64M
  const size_t nQKV = (size_t)S_LEN * 3 * E_DIM;      // 11.8M
  const size_t nQK  = (size_t)H_NUM * S_LEN * DP;     // 4.72M

  bf16_t* x_hi  = (bf16_t*)carve(nX * 2);
  bf16_t* x_lo  = (bf16_t*)carve(nX * 2);
  bf16_t* w1_hi = (bf16_t*)carve(nW1 * 2);
  bf16_t* w1_lo = (bf16_t*)carve(nW1 * 2);
  bf16_t* w2_hi = (bf16_t*)carve(nW2 * 2);
  bf16_t* w2_lo = (bf16_t*)carve(nW2 * 2);
  float*  qkv   = (float*) carve(nQKV * 4);
  bf16_t* q_hi  = (bf16_t*)carve(nQK * 2);
  bf16_t* q_lo  = (bf16_t*)carve(nQK * 2);
  bf16_t* k_hi  = (bf16_t*)carve(nQK * 2);
  bf16_t* k_lo  = (bf16_t*)carve(nQK * 2);
  bf16_t* vt_hi = (bf16_t*)carve(nQK * 2);
  bf16_t* vt_lo = (bf16_t*)carve(nQK * 2);
  float*  ctx    = qkv;    // reuse: qkv dead after RoPE
  bf16_t* ctx_hi = x_hi;   // reuse: x splits dead after QKV GEMM
  bf16_t* ctx_lo = x_lo;

  // 1) split inputs to bf16 hi/lo
  split_bf16x2_kernel<<<(nX + 255) / 256, 256, 0, stream>>>(x, x_hi, x_lo, (int)nX);
  split_bf16x2_kernel<<<(nW1 + 255) / 256, 256, 0, stream>>>(qkv_w, w1_hi, w1_lo, (int)nW1);
  split_bf16x2_kernel<<<(nW2 + 255) / 256, 256, 0, stream>>>(proj_w, w2_hi, w2_lo, (int)nW2);

  // 2) QKV GEMM: [3072,1280] @ [3840,1280]^T + b  -> qkv f32
  gemm_bf16x3_kernel<<<dim3((3 * E_DIM) / 256, S_LEN / 64), 256, 0, stream>>>(
      x_hi, x_lo, w1_hi, w1_lo, qkv_b, qkv, S_LEN, 3 * E_DIM, E_DIM);

  // 3) RoPE + split + pad + v-transpose
  {
    size_t n = (size_t)S_LEN * H_NUM * DP;
    rope_split_kernel<<<(n + 255) / 256, 256, 0, stream>>>(
        qkv, cosb, sinb, q_hi, q_lo, k_hi, k_lo, vt_hi, vt_lo);
  }

  // 4) segment-masked flash attention -> ctx f32 [S][E]
  {
    int waves = H_NUM * (S_LEN / 32);
    attn_flash_kernel<<<waves / ATT_WAVES, 32 * ATT_WAVES, 0, stream>>>(
        q_hi, q_lo, k_hi, k_lo, vt_hi, vt_lo, cu, ctx);
  }

  // 5) split ctx, then output projection -> d_out
  split_bf16x2_kernel<<<(nX + 255) / 256, 256, 0, stream>>>(ctx, ctx_hi, ctx_lo, (int)nX);
  gemm_bf16x3_kernel<<<dim3(E_DIM / 256, S_LEN / 64), 256, 0, stream>>>(
      ctx_hi, ctx_lo, w2_hi, w2_lo, proj_b, out, S_LEN, E_DIM, E_DIM);
}
